// Encoder_77653008712116
// MI455X (gfx1250) — matrix operand order repaired and checked by reading, not compiled, not run on hardware
//
#include <hip/hip_runtime.h>

typedef __attribute__((ext_vector_type(16))) _Float16 v16h;
typedef __attribute__((ext_vector_type(8)))  _Float16 v8h;
typedef __attribute__((ext_vector_type(8)))  float    v8f;

__device__ __forceinline__ void atomAddF(float* p, float v) {
  __hip_atomic_fetch_add(p, v, __ATOMIC_RELAXED, __HIP_MEMORY_SCOPE_AGENT);
}

// ------------------------------------------------ zero (n+1)*C floats, n from device scalar
__global__ __launch_bounds__(256)
void zero_acc_k(float* __restrict__ p, const int* nPtr, int nConst, int C) {
  int  n = nPtr ? *nPtr : nConst;
  long i = (long)blockIdx.x * 256 + threadIdx.x;
  if (i < (long)(n + 1) * C) p[i] = 0.0f;
}

// ------------------------------------------------ c0: Cin=1 -> Cout=64 (no WMMA: 1-wide K)
__global__ __launch_bounds__(256)
void conv_c0_k(const float* __restrict__ feats, const float* __restrict__ w,
               const int* __restrict__ inMap, const int* __restrict__ outMap,
               float* __restrict__ outAcc, int n0, int M) {
  long g = (long)blockIdx.x * 256 + threadIdx.x;
  long pair = g >> 6;
  int  c    = (int)(g & 63);
  if (pair >= (long)9 * M) return;
  int k = (int)(pair / M);
  int iidx = inMap[pair];
  float f = (iidx < n0) ? feats[iidx] : 0.0f;   // dummy input row -> 0
  float v = f * w[k * 64 + c];
  atomAddF(outAcc + (long)outMap[pair] * 64 + c, v);
}

// ------------------------------------------------ WMMA sparse conv (f16 in, f32 acc)
// Block owns offset k. W_k staged to LDS transposed once; B fragments hoisted to
// registers. 32-row pair tiles gathered with GLOBAL_LOAD_ASYNC_TO_LDS_B128 into
// a double-buffered LDS tile (ASYNCcnt pipelined against the previous tile's
// WMMAs), two 16x16 slabs per wave via v_wmma_f32_16x16x32_f16, scatter-add
// with global_atomic_add_f32.
template <int CIN>
__global__ __launch_bounds__(256)
void conv_wmma_k(const _Float16* __restrict__ feats, // (nIn+1) x CIN f16, row nIn == 0
                 const float* __restrict__ w,        // (9, CIN, 128) f32
                 const int* __restrict__ inMap,      // (9, M)
                 const int* __restrict__ outMap,     // (9, M)
                 float* __restrict__ outAcc,         // (nOut+1) x 128 f32, pre-zeroed
                 const int* __restrict__ nInPtr, const int* __restrict__ nOutPtr,
                 int M, int nblkPerK) {
  constexpr int COUT = 128;
  constexpr int AS   = CIN + 8;         // padded LDS stride (f16 elems)
  constexpr int KB   = CIN / 32;
  constexpr int CPR  = CIN / 8;         // 16B chunks per row
  constexpr int ABUF = 32 * AS;         // one A tile (f16 elems)
  __shared__ __attribute__((aligned(16))) _Float16 ldsW[COUT * AS]; // wT[col][k]
  __shared__ __attribute__((aligned(16))) _Float16 ldsA[2 * ABUF];  // ping-pong tiles

  const int tid  = threadIdx.x;
  const int k    = blockIdx.x / nblkPerK;
  const int blk  = blockIdx.x % nblkPerK;
  const int nIn  = *nInPtr;
  const int nOut = *nOutPtr;

  const float* wk = w + (long)k * CIN * COUT;
  for (int i = tid; i < CIN * COUT; i += 256) {      // stage transposed f16 weights
    int ci = i >> 7;                                 // / COUT
    int co = i & 127;
    ldsW[co * AS + ci] = (_Float16)wk[i];
  }
  __syncthreads();

  const int lane   = tid & 31;
  const int wv     = tid >> 5;                       // wave -> 16-column slab
  const int laneHi = lane >> 4;
  const int ln     = lane & 15;
  const int col    = wv * 16 + ln;

  // Hoist B fragments (block-invariant) into registers: K 8h..8h+7 and 16+8h..16+8h+7 per lane.
  v16h Bf[KB];
#pragma unroll
  for (int kb = 0; kb < KB; ++kb) {
    const _Float16* bp = ldsW + col * AS + kb * 32 + laneHi * 8;
    v8h b0 = *(const v8h*)(bp);
    v8h b1 = *(const v8h*)(bp + 16);
    Bf[kb] = __builtin_shufflevector(b0, b1, 0,1,2,3,4,5,6,7,8,9,10,11,12,13,14,15);
  }

  const int* inK  = inMap  + (long)k * M;
  const int* outK = outMap + (long)k * M;
  const int ntiles = (M + 31) >> 5;

  // Per-lane async global->LDS gather of one 32-row f16 tile (all lanes active).
  auto gatherAsync = [&](int t, _Float16* dst) {
    const int rbase = t << 5;
#pragma unroll
    for (int i = tid; i < 32 * CPR; i += 256) {
      int r = i / CPR, c = i % CPR;
      int rowp = rbase + r;
      int idx  = (rowp < M) ? inK[rowp] : nIn;       // tail rows -> dummy zero row
      const _Float16* src = feats + (long)idx * CIN + c * 8;
      unsigned ldsOff = (unsigned)(size_t)(dst + r * AS + c * 8); // flat low 32b = LDS addr
      asm volatile("global_load_async_to_lds_b128 %0, %1, off"
                   :: "v"(ldsOff), "v"(src)
                   : "memory");
    }
  };

  if (blk < ntiles) gatherAsync(blk, ldsA);          // prime the pipeline
  int pb = 0;
  for (int t = blk; t < ntiles; t += nblkPerK, pb ^= 1) {
    asm volatile("s_wait_asynccnt 0x0" ::: "memory"); // own tile writes landed
    __syncthreads();                                  // everyone's writes visible
    const int tn = t + nblkPerK;
    if (tn < ntiles) gatherAsync(tn, ldsA + (pb ^ 1) * ABUF); // prefetch next tile
    const _Float16* cur = ldsA + pb * ABUF;
    const int rbase = t << 5;

    v8f acc0 = {}, acc1 = {};
#pragma unroll
    for (int kb = 0; kb < KB; ++kb) {
      const _Float16* ap0 = cur + ln * AS + kb * 32 + laneHi * 8;
      v8h x0 = *(const v8h*)(ap0);
      v8h x1 = *(const v8h*)(ap0 + 16);
      v16h a0 = __builtin_shufflevector(x0, x1, 0,1,2,3,4,5,6,7,8,9,10,11,12,13,14,15);
      acc0 = __builtin_amdgcn_wmma_f32_16x16x32_f16(false, a0, false, Bf[kb],
                                                    (short)0, acc0, false, false);
      const _Float16* ap1 = cur + (16 + ln) * AS + kb * 32 + laneHi * 8;
      v8h y0 = *(const v8h*)(ap1);
      v8h y1 = *(const v8h*)(ap1 + 16);
      v16h a1 = __builtin_shufflevector(y0, y1, 0,1,2,3,4,5,6,7,8,9,10,11,12,13,14,15);
      acc1 = __builtin_amdgcn_wmma_f32_16x16x32_f16(false, a1, false, Bf[kb],
                                                    (short)0, acc1, false, false);
    }

#pragma unroll
    for (int r = 0; r < 8; ++r) {                    // C layout: VGPR r -> row r + 8*hi
      int m = laneHi * 8 + r;
      int rowp0 = rbase + m;
      int oidx0 = (rowp0 < M) ? outK[rowp0] : nOut;
      atomAddF(outAcc + (long)oidx0 * COUT + col, acc0[r]);
      int rowp1 = rbase + 16 + m;
      int oidx1 = (rowp1 < M) ? outK[rowp1] : nOut;
      atomAddF(outAcc + (long)oidx1 * COUT + col, acc1[r]);
    }
  }
}

// ------------------------------------------------ bias (+relu) epilogue
// f32 acc -> f16 feature buffer (next layer) + optional f32 d_out region at
// 128*(*off1 + *off2). Zeroes the dummy feature row n.
__global__ __launch_bounds__(256)
void epilogue_k(const float* __restrict__ acc, const float* __restrict__ bias,
                _Float16* __restrict__ featDst, float* __restrict__ dout,
                const int* nPtr, int nConst, const int* off1, const int* off2,
                int C, int relu) {
  int  n   = nPtr ? *nPtr : nConst;
  long i   = (long)blockIdx.x * 256 + threadIdx.x;
  long row = i / C;
  int  c   = (int)(i % C);
  if (row > n) return;
  if (row == n) { if (featDst) featDst[i] = (_Float16)0.0f; return; }
  float v = acc[i] + bias[c];
  if (relu && v < 0.0f) v = 0.0f;
  if (featDst) featDst[i] = (_Float16)v;
  if (dout) {
    long off = 0;
    if (off1) off += *off1;
    if (off2) off += *off2;
    dout[off * C + i] = v;
  }
}

// ------------------------------------------------ host driver
extern "C" void kernel_launch(void* const* d_in, const int* in_sizes, int n_in,
                              void* d_out, int out_size, void* d_ws, size_t ws_size,
                              hipStream_t stream) {
  const float* feats = (const float*)d_in[0];
  const float* w_c0 = (const float*)d_in[1];  const float* b_c0 = (const float*)d_in[2];
  const int* in_c0 = (const int*)d_in[3];     const int* out_c0 = (const int*)d_in[4];
  const float* w_d0 = (const float*)d_in[5];  const float* b_d0 = (const float*)d_in[6];
  const int* in_d0 = (const int*)d_in[7];     const int* out_d0 = (const int*)d_in[8];
  const float* w_c1 = (const float*)d_in[9];  const float* b_c1 = (const float*)d_in[10];
  const int* in_c1 = (const int*)d_in[11];    const int* out_c1 = (const int*)d_in[12];
  const float* w_d1 = (const float*)d_in[13]; const float* b_d1 = (const float*)d_in[14];
  const int* in_d1 = (const int*)d_in[15];    const int* out_d1 = (const int*)d_in[16];
  const float* w_c2 = (const float*)d_in[17]; const float* b_c2 = (const float*)d_in[18];
  const int* in_c2 = (const int*)d_in[19];    const int* out_c2 = (const int*)d_in[20];
  const float* w_d2 = (const float*)d_in[21]; const float* b_d2 = (const float*)d_in[22];
  const int* in_d2 = (const int*)d_in[23];    const int* out_d2 = (const int*)d_in[24];
  const int* n0p = (const int*)d_in[25];
  const int* n1p = (const int*)d_in[26];
  const int* n2p = (const int*)d_in[27];
  const int* n3p = (const int*)d_in[28];

  const int n0  = in_sizes[0];
  const int Mc0 = in_sizes[3] / 9,  Md0 = in_sizes[7] / 9;
  const int Mc1 = in_sizes[11] / 9, Md1 = in_sizes[15] / 9;
  const int Mc2 = in_sizes[19] / 9, Md2 = in_sizes[23] / 9;

  // Hard caps on pyramid level sizes (grids 512^2 / 256^2 / 128^2).
  const int CAP1 = n0 < 262144 ? n0 : 262144;
  const int CAP2 = n0 < 65536  ? n0 : 65536;
  const int CAP3 = n0 < 16384  ? n0 : 16384;

  // Workspace: one shared f32 accumulator + f16 feature buffers (256B aligned).
  char* base = (char*)d_ws;
  auto carve = [&](size_t bytes) {
    char* p = base; base += (bytes + 255) & ~(size_t)255; return p;
  };
  size_t accElems = (size_t)(n0 + 1) * 64;
  if ((size_t)(CAP1 + 1) * 128 > accElems) accElems = (size_t)(CAP1 + 1) * 128;
  float*     acc = (float*)carve(accElems * 4);
  _Float16*  f0  = (_Float16*)carve((size_t)(n0 + 1) * 64 * 2);    // level-0 feats
  _Float16*  f1  = (_Float16*)carve((size_t)(CAP1 + 1) * 128 * 2); // out0
  _Float16*  f2  = (_Float16*)carve((size_t)(CAP1 + 1) * 128 * 2); // y, later z
  _Float16*  f3  = (_Float16*)carve((size_t)(CAP2 + 1) * 128 * 2); // out1
  float*     dout = (float*)d_out;

  auto zeroAcc = [&](const int* np, int nc, int C, int capRows) {
    long total = (long)(capRows + 1) * C;
    zero_acc_k<<<dim3((unsigned)((total + 255) / 256)), dim3(256), 0, stream>>>(
        acc, np, nc, C);
  };
  auto epi = [&](const float* bias, _Float16* fdst, float* dst,
                 const int* np, int nc, const int* o1, const int* o2,
                 int C, int relu, int capRows) {
    long total = (long)(capRows + 1) * C;
    epilogue_k<<<dim3((unsigned)((total + 255) / 256)), dim3(256), 0, stream>>>(
        acc, bias, fdst, dst, np, nc, o1, o2, C, relu);
  };
  auto nbFor = [](int M) { int t = (M + 31) / 32; if (t < 1) t = 1; return t < 96 ? t : 96; };

  // x = conv_c0(feats) + b          -> f0
  zeroAcc(nullptr, n0, 64, n0);
  {
    long work = (long)9 * Mc0 * 64;
    conv_c0_k<<<dim3((unsigned)((work + 255) / 256)), dim3(256), 0, stream>>>(
        feats, w_c0, in_c0, out_c0, acc, n0, Mc0);
  }
  epi(b_c0, f0, nullptr, nullptr, n0, nullptr, nullptr, 64, 0, n0);

  // out0 = relu(conv_d0(x) + b)     -> f1, d_out @ 128*(n2+n3)
  zeroAcc(n1p, 0, 128, CAP1);
  { int nb = nbFor(Md0);
    conv_wmma_k<64><<<dim3(9 * nb), dim3(256), 0, stream>>>(
        f0, w_d0, in_d0, out_d0, acc, n0p, n1p, Md0, nb); }
  epi(b_d0, f1, dout, n1p, 0, n2p, n3p, 128, 1, CAP1);

  // y = conv_c1(out0) + b           -> f2
  zeroAcc(n1p, 0, 128, CAP1);
  { int nb = nbFor(Mc1);
    conv_wmma_k<128><<<dim3(9 * nb), dim3(256), 0, stream>>>(
        f1, w_c1, in_c1, out_c1, acc, n1p, n1p, Mc1, nb); }
  epi(b_c1, f2, nullptr, n1p, 0, nullptr, nullptr, 128, 0, CAP1);

  // out1 = relu(conv_d1(y) + b)     -> f3, d_out @ 128*n3
  zeroAcc(n2p, 0, 128, CAP2);
  { int nb = nbFor(Md1);
    conv_wmma_k<128><<<dim3(9 * nb), dim3(256), 0, stream>>>(
        f2, w_d1, in_d1, out_d1, acc, n1p, n2p, Md1, nb); }
  epi(b_d1, f3, dout, n2p, 0, n3p, nullptr, 128, 1, CAP2);

  // z = conv_c2(out1) + b           -> f2 (reuse)
  zeroAcc(n2p, 0, 128, CAP2);
  { int nb = nbFor(Mc2);
    conv_wmma_k<128><<<dim3(9 * nb), dim3(256), 0, stream>>>(
        f3, w_c2, in_c2, out_c2, acc, n2p, n2p, Mc2, nb); }
  epi(b_c2, f2, nullptr, n2p, 0, nullptr, nullptr, 128, 0, CAP2);

  // out2 = conv_d2(z) + b           -> d_out @ 0 (no relu)
  zeroAcc(n3p, 0, 128, CAP3);
  { int nb = nbFor(Md2);
    conv_wmma_k<128><<<dim3(9 * nb), dim3(256), 0, stream>>>(
        f2, w_d2, in_d2, out_d2, acc, n2p, n3p, Md2, nb); }
  epi(b_d2, nullptr, dout, n3p, 0, nullptr, nullptr, 128, 0, CAP3);
}
